// BulkFormerGAT_59270548685255
// MI455X (gfx1250) — matrix operand order repaired and checked by reading, not compiled, not run on hardware
//
#include <hip/hip_runtime.h>
#include <cmath>

// ---------------------------------------------------------------------------
// BulkFormer-GAT forward for MI455X (gfx1250), wave32 + WMMA f16 16x16x32.
//
// GEMM: 128x64 block tile, 8 wave32s each owning 32x32 (4 WMMAs/K-step),
// double-buffered LDS, software-pipelined GLOBAL_LOAD_ASYNC_TO_LDS_B128
// staging (ASYNCcnt): issue tile k+1, compute tile k, wait, one barrier.
// Operands pre-converted to f16 (weights pre-transposed to [N][K], ragged K
// zero-padded to %32==0) so staging is a pure aligned 16B-per-lane memcpy.
// ---------------------------------------------------------------------------

#define DIMC 320
#define HC   4
#define DHC  80
#define GC   19357
#define NBC  350
#define BC   2
#define EC   160000
#define ETOT (EC + GC)
#define FFC  (4 * DIMC)            // 1280
#define NRC  ((long long)BC * GC)  // 38714
#define KG   19360                 // G padded to %32
#define KDH  96                    // DH=80 padded to %32
#define KNB  352                   // NB=350 padded to %32

#define USE_ASYNC_LDS 1

typedef __attribute__((ext_vector_type(16))) _Float16 v16h;
typedef __attribute__((ext_vector_type(8)))  _Float16 v8h;
typedef __attribute__((ext_vector_type(8)))  float    v8f;

// ------------------------------ small helpers ------------------------------

__device__ __forceinline__ float warp_sum(float v) {
  #pragma unroll
  for (int o = 16; o; o >>= 1) v += __shfl_xor(v, o, 32);
  return v;
}
__device__ __forceinline__ float warp_max(float v) {
  #pragma unroll
  for (int o = 16; o; o >>= 1) v = fmaxf(v, __shfl_xor(v, o, 32));
  return v;
}
__device__ __forceinline__ void atomicMaxF(float* addr, float v) {
  if (v >= 0.0f) atomicMax((int*)addr, __float_as_int(v));
  else           atomicMin((unsigned int*)addr, __float_as_uint(v));
}

__global__ void fill_kernel(float* __restrict__ p, float v, long long n) {
  long long t = blockIdx.x * 256LL + threadIdx.x;
  if (t < n) p[t] = v;
}

// dst[r*ldd + c] = (c < C) ? src[r*lds + c]*scale : 0    (r<R, c<ldd)
__global__ void cvt16_rows_kernel(const float* __restrict__ src, long long lds,
                                  _Float16* __restrict__ dst, long long ldd,
                                  long long R, long long C, float scale) {
  long long t = blockIdx.x * 256LL + threadIdx.x;
  if (t >= R * ldd) return;
  long long r = t / ldd, c = t % ldd;
  float v = (c < C) ? src[r * lds + c] * scale : 0.0f;
  dst[t] = (_Float16)v;
}

// dst[c*ldd + r] = (r < R) ? src[r*lds + c]*scale : 0    (c<Cdst, r<ldd)
__global__ void cvt16_trans_kernel(const float* __restrict__ src, long long lds,
                                   _Float16* __restrict__ dst, long long ldd,
                                   long long R, long long Cdst, float scale) {
  long long t = blockIdx.x * 256LL + threadIdx.x;
  if (t >= Cdst * ldd) return;
  long long c = t / ldd, r = t % ldd;
  float v = (r < R) ? src[r * lds + c] * scale : 0.0f;
  dst[t] = (_Float16)v;
}

// ------------------------------ WMMA GEMM ----------------------------------
// C[M,N] (f32 or f16) = act(A[M,K]f16 @ Wt[N,K]^T f16 + bias)*row_scale + res
// Preconditions: K%32==0, lda/ldw rows 16B aligned, lda>=K, ldw>=K.
__global__ void __launch_bounds__(256)
wmma_gemm_f16(const _Float16* __restrict__ A, long long lda,
              const _Float16* __restrict__ Wt, long long ldw,
              const float* __restrict__ bias,
              const float* __restrict__ residual, long long ldr,
              const float* __restrict__ row_scale,
              float* __restrict__ C, _Float16* __restrict__ C16, long long ldc,
              int M, int N, int K, int act) {
  __shared__ __align__(16) _Float16 As[2][128 * 32];   // [m][k], double buffered
  __shared__ __align__(16) _Float16 Ws[2][64 * 32];    // [n][k], double buffered

  const int tid  = threadIdx.x;
  const int lane = tid & 31;
  const int wave = tid >> 5;
  const int mw   = wave & 3;          // 4 M-strips of 32
  const int nw   = wave >> 2;         // 2 N-strips of 32
  const int bm   = blockIdx.x * 128;
  const int bn   = blockIdx.y * 64;
  const int hi   = lane >> 4;
  const int nl   = lane & 15;
  const bool interior = (bm + 128 <= M) && (bn + 64 <= N);  // block-uniform

  // Stage one 128x32 A tile (2 chunks/thread) + 64x32 W tile (1 chunk/thread).
  auto stage = [&](int kk, int buf) {
    #pragma unroll
    for (int ch = 0; ch < 2; ++ch) {
      int cidx = tid + ch * 256;
      int r = cidx >> 2, seg = (cidx & 3) * 8;
      const _Float16* ga = A + (long long)(bm + r) * lda + kk + seg;
      _Float16* la = &As[buf][r * 32 + seg];
#if USE_ASYNC_LDS
      if (interior) {
        asm volatile("global_load_async_to_lds_b128 %0, %1, off"
                     :: "v"((unsigned)(uintptr_t)la), "v"(ga) : "memory");
      } else
#endif
      {
        v8h z = {};
        *(v8h*)la = (bm + r < M) ? *(const v8h*)ga : z;
      }
    }
    {
      int r = tid >> 2, seg = (tid & 3) * 8;
      const _Float16* gw = Wt + (long long)(bn + r) * ldw + kk + seg;
      _Float16* lw = &Ws[buf][r * 32 + seg];
#if USE_ASYNC_LDS
      if (interior) {
        asm volatile("global_load_async_to_lds_b128 %0, %1, off"
                     :: "v"((unsigned)(uintptr_t)lw), "v"(gw) : "memory");
      } else
#endif
      {
        v8h z = {};
        *(v8h*)lw = (bn + r < N) ? *(const v8h*)gw : z;
      }
    }
  };

  // Fragment loads per ISA 7.12.2 16-bit layouts.
  auto load_a = [&](const _Float16* base, int mrow) {
    v16h a;
    const _Float16* p = base + (mrow + nl) * 32;
    #pragma unroll
    for (int e = 0; e < 8; ++e) a[e] = p[8 * hi + e];
    #pragma unroll
    for (int e = 0; e < 8; ++e) a[8 + e] = p[16 + 8 * hi + e];
    return a;
  };
  auto load_b = [&](const _Float16* base, int ncol) {
    v16h bv;
    const _Float16* p = base + (ncol + nl) * 32 + 8 * hi;
    #pragma unroll
    for (int e = 0; e < 8; ++e) bv[e] = p[e];
    #pragma unroll
    for (int e = 0; e < 8; ++e) bv[8 + e] = p[16 + e];
    return bv;
  };

  v8f acc00 = {}, acc01 = {}, acc10 = {}, acc11 = {};

  stage(0, 0);
#if USE_ASYNC_LDS
  if (interior) asm volatile("s_wait_asynccnt 0x0" ::: "memory");
#endif
  __syncthreads();

  const int ntiles = K >> 5;
  for (int i = 0; i < ntiles; ++i) {
    const int buf = i & 1;
    if (i + 1 < ntiles) stage((i + 1) << 5, buf ^ 1);   // async prefetch next tile

    const _Float16* ab = As[buf];
    const _Float16* wb = Ws[buf];
    v16h a0 = load_a(ab, mw * 32);
    v16h a1 = load_a(ab, mw * 32 + 16);
    v16h b0 = load_b(wb, nw * 32);
    v16h b1 = load_b(wb, nw * 32 + 16);
    acc00 = __builtin_amdgcn_wmma_f32_16x16x32_f16(false, a0, false, b0, (short)0, acc00, false, false);
    acc01 = __builtin_amdgcn_wmma_f32_16x16x32_f16(false, a0, false, b1, (short)0, acc01, false, false);
    acc10 = __builtin_amdgcn_wmma_f32_16x16x32_f16(false, a1, false, b0, (short)0, acc10, false, false);
    acc11 = __builtin_amdgcn_wmma_f32_16x16x32_f16(false, a1, false, b1, (short)0, acc11, false, false);

#if USE_ASYNC_LDS
    if (interior) asm volatile("s_wait_asynccnt 0x0" ::: "memory");
#endif
    __syncthreads();
  }

  // Epilogue: C/D layout -> lane = N col, VGPR rr -> M = rr + 8*(lane>=16).
  auto emit = [&](int gm, int gn, float v) {
    if (gm >= M || gn >= N) return;
    if (bias) v += bias[gn];
    if (act == 1) v = 0.5f * v * (1.0f + erff(v * 0.70710678118654752f));
    if (row_scale) v *= row_scale[gm];
    if (residual) v += residual[(long long)gm * ldr + gn];
    if (C16) C16[(long long)gm * ldc + gn] = (_Float16)v;
    else     C[(long long)gm * ldc + gn] = v;
  };
  #pragma unroll
  for (int rr = 0; rr < 8; ++rr) {
    int gm0 = bm + mw * 32 + hi * 8 + rr;
    int gn0 = bn + nw * 32 + nl;
    emit(gm0,      gn0,      acc00[rr]);
    emit(gm0,      gn0 + 16, acc01[rr]);
    emit(gm0 + 16, gn0,      acc10[rr]);
    emit(gm0 + 16, gn0 + 16, acc11[rr]);
  }
}

// ------------------------- pointwise / reduction kernels -------------------

__global__ void ctx_kernel(const int* __restrict__ idx, const float* __restrict__ ae_latent,
                           const float* __restrict__ ae_w, const float* __restrict__ ae_b,
                           float* __restrict__ ctx) {
  int b = blockIdx.x, c = threadIdx.x;
  const float* a = ae_latent + (long long)idx[b] * DIMC;
  float s = ae_b[c];
  for (int k = 0; k < DIMC; ++k) s += a[k] * ae_w[(long long)k * DIMC + c];
  ctx[b * DIMC + c] = s;
}

__global__ void build_pre_kernel(const float* __restrict__ x_expr, const float* __restrict__ gene,
                                 const float* __restrict__ ctx, float* __restrict__ pre) {
  long long t = blockIdx.x * 256LL + threadIdx.x;
  if (t >= NRC * DIMC) return;
  int c = (int)(t % DIMC);
  long long bg = t / DIMC;
  int g = (int)(bg % GC);
  int b = (int)(bg / GC);
  float xv = x_expr[(long long)b * GC + g];
  int i = (c < DIMC / 2) ? c : c - DIMC / 2;
  float theta = powf(100.0f, 2.0f * (float)(i + 1) / (float)DIMC);
  float ang = xv * theta;
  float rv = (c < DIMC / 2) ? sinf(ang) : cosf(ang);
  pre[t] = rv + gene[(long long)g * DIMC + c] + ctx[b * DIMC + c];
}

__global__ void edge_logit_kernel(const float* __restrict__ xl, const float* __restrict__ xr,
                                  const float* __restrict__ att, const int* __restrict__ eidx,
                                  float* __restrict__ logit, float* __restrict__ lmax) {
  long long t = blockIdx.x * 256LL + threadIdx.x;
  if (t >= (long long)BC * ETOT * HC) return;
  int h = (int)(t % HC);
  long long eb = t / HC;
  int e = (int)(eb % ETOT);
  int b = (int)(eb / ETOT);
  int s, d;
  if (e < EC) { s = eidx[e]; d = eidx[EC + e]; } else { s = e - EC; d = s; }
  const float* pl = xl + ((long long)b * GC + s) * DIMC + h * DHC;
  const float* pr = xr + ((long long)b * GC + d) * DIMC + h * DHC;
  const float* pa = att + h * DHC;
  float acc = 0.0f;
  for (int k = 0; k < DHC; ++k) {
    float v = pl[k] + pr[k];
    v = v > 0.0f ? v : 0.2f * v;
    acc += v * pa[k];
  }
  logit[((long long)b * ETOT + e) * HC + h] = acc;
  atomicMaxF(&lmax[((long long)b * GC + d) * HC + h], acc);
}

__global__ void edge_exp_kernel(const float* __restrict__ logit, const float* __restrict__ lmax,
                                const int* __restrict__ eidx, float* __restrict__ ex,
                                float* __restrict__ den) {
  long long t = blockIdx.x * 256LL + threadIdx.x;
  if (t >= (long long)BC * ETOT * HC) return;
  int h = (int)(t % HC);
  long long eb = t / HC;
  int e = (int)(eb % ETOT);
  int b = (int)(eb / ETOT);
  int d = (e < EC) ? eidx[EC + e] : (e - EC);
  float v = expf(logit[t] - lmax[((long long)b * GC + d) * HC + h]);
  ex[t] = v;
  atomicAdd(&den[((long long)b * GC + d) * HC + h], v);
}

__global__ void edge_aggr_kernel(const float* __restrict__ ex, const float* __restrict__ den,
                                 const float* __restrict__ xl, const int* __restrict__ eidx,
                                 float* __restrict__ out) {
  long long t = blockIdx.x * 256LL + threadIdx.x;
  if (t >= (long long)BC * ETOT * DIMC) return;
  int c = (int)(t % DIMC);
  long long eb = t / DIMC;
  int e = (int)(eb % ETOT);
  int b = (int)(eb / ETOT);
  int s, d;
  if (e < EC) { s = eidx[e]; d = eidx[EC + e]; } else { s = e - EC; d = s; }
  int h = c / DHC;
  float alpha = ex[((long long)b * ETOT + e) * HC + h] /
                (den[((long long)b * GC + d) * HC + h] + 1e-16f);
  atomicAdd(&out[((long long)b * GC + d) * DIMC + c],
            alpha * xl[((long long)b * GC + s) * DIMC + c]);
}

__global__ void add_bias_kernel(const float* __restrict__ in, const float* __restrict__ bias,
                                float* __restrict__ out) {
  long long t = blockIdx.x * 256LL + threadIdx.x;
  if (t >= NRC * DIMC) return;
  out[t] = in[t] + bias[t % DIMC];
}

__global__ void layernorm_kernel(const float* __restrict__ x, const float* __restrict__ g,
                                 const float* __restrict__ b, float* __restrict__ y,
                                 long long rows) {
  int lane = threadIdx.x & 31;
  long long row = blockIdx.x * 8LL + (threadIdx.x >> 5);
  if (row >= rows) return;
  const float* p = x + row * DIMC;
  float s = 0.0f;
  for (int i = lane; i < DIMC; i += 32) s += p[i];
  s = warp_sum(s);
  float mu = s / (float)DIMC;
  float v = 0.0f;
  for (int i = lane; i < DIMC; i += 32) { float d = p[i] - mu; v += d * d; }
  v = warp_sum(v);
  float inv = rsqrtf(v / (float)DIMC + 1e-5f);
  for (int i = lane; i < DIMC; i += 32)
    y[row * DIMC + i] = (p[i] - mu) * inv * g[i] + b[i];
}

__global__ void diag_kernel(const float* __restrict__ slice, float* __restrict__ diag, float scale) {
  int n = blockIdx.x * 256 + threadIdx.x;
  if (n >= GC) return;
  const float* p = slice + (long long)n * DIMC;
  float s = 0.0f;
  for (int d = 0; d < DHC; ++d) { float v = p[d]; s += v * v; }
  diag[n] = s * scale;
}

__global__ void reduce_max_kernel(const float* __restrict__ x, long long n, float* __restrict__ out) {
  float m = -1e30f;
  for (long long i = blockIdx.x * (long long)blockDim.x + threadIdx.x; i < n;
       i += (long long)gridDim.x * blockDim.x)
    m = fmaxf(m, x[i]);
  m = warp_max(m);
  if ((threadIdx.x & 31) == 0) atomicMaxF(out, m);
}

__global__ void rowmax_kernel(const float* __restrict__ dd, float* __restrict__ rmax) {
  int lane = threadIdx.x & 31;
  long long row = blockIdx.x * 8LL + (threadIdx.x >> 5);
  if (row >= GC) return;
  const float* p = dd + row * (long long)NBC;
  float m = -1e30f;
  for (int i = lane; i < NBC; i += 32) m = fmaxf(m, p[i]);
  m = warp_max(m);
  if (lane == 0) rmax[row] = m;
}

__global__ void favor_exp_kernel(float* __restrict__ dd, const float* __restrict__ diag,
                                 const float* __restrict__ mx, int per_row) {
  long long t = blockIdx.x * 256LL + threadIdx.x;
  if (t >= (long long)GC * NBC) return;
  long long n = t / NBC;
  float m = per_row ? mx[n] : mx[0];
  const float ratio = 0.05345224838248488f;  // 350^-0.5
  dd[t] = ratio * (expf(dd[t] - diag[n] - m) + 1e-4f);
}

__global__ void colsum_kernel(const float* __restrict__ kp, float* __restrict__ ksum) {
  int m = threadIdx.x;
  if (m >= NBC) return;
  long long n0 = blockIdx.x * 128LL;
  long long n1 = n0 + 128 < GC ? n0 + 128 : GC;
  float s = 0.0f;
  for (long long n = n0; n < n1; ++n) s += kp[n * NBC + m];
  atomicAdd(&ksum[m], s);
}

__global__ void dinv_kernel(const float* __restrict__ qp, const float* __restrict__ ksum,
                            float* __restrict__ dinv) {
  int lane = threadIdx.x & 31;
  long long row = blockIdx.x * 8LL + (threadIdx.x >> 5);
  if (row >= GC) return;
  const float* p = qp + row * (long long)NBC;
  float s = 0.0f;
  for (int i = lane; i < NBC; i += 32) s += p[i] * ksum[i];
  s = warp_sum(s);
  if (lane == 0) dinv[row] = 1.0f / s;
}

__global__ void head_kernel(const float* __restrict__ x, const float* __restrict__ hw,
                            const float* __restrict__ hb, float* __restrict__ out, long long rows) {
  int lane = threadIdx.x & 31;
  long long row = blockIdx.x * 8LL + (threadIdx.x >> 5);
  if (row >= rows) return;
  const float* p = x + row * DIMC;
  float s = 0.0f;
  for (int i = lane; i < DIMC; i += 32) s += p[i] * hw[i];
  s = warp_sum(s);
  if (lane == 0) out[row] = s + hb[0];
}

// ------------------------------ orchestration ------------------------------

extern "C" void kernel_launch(void* const* d_in, const int* in_sizes, int n_in,
                              void* d_out, int out_size, void* d_ws, size_t ws_size,
                              hipStream_t stream) {
  (void)in_sizes; (void)n_in; (void)out_size; (void)ws_size;

  const int*   idx       = (const int*)  d_in[0];
  const float* x_expr    = (const float*)d_in[1];
  const float* esm2      = (const float*)d_in[2];
  const float* ae_latent = (const float*)d_in[3];
  const int*   eidx      = (const int*)  d_in[4];
  const float* gene_w    = (const float*)d_in[5];
  const float* gene_b    = (const float*)d_in[6];
  const float* ae_w      = (const float*)d_in[7];
  const float* ae_b      = (const float*)d_in[8];
  const float* expr_w    = (const float*)d_in[9];
  const float* expr_b    = (const float*)d_in[10];
  const float* gat_wl    = (const float*)d_in[11];
  const float* gat_bl    = (const float*)d_in[12];
  const float* gat_wr    = (const float*)d_in[13];
  const float* gat_br    = (const float*)d_in[14];
  const float* gat_att   = (const float*)d_in[15];
  const float* gat_bias  = (const float*)d_in[16];
  const float* ln1g      = (const float*)d_in[17];
  const float* ln1b      = (const float*)d_in[18];
  const float* wq        = (const float*)d_in[19];
  const float* wk        = (const float*)d_in[20];
  const float* wv        = (const float*)d_in[21];
  const float* wo        = (const float*)d_in[22];
  const float* bo        = (const float*)d_in[23];
  const float* proj      = (const float*)d_in[24];
  const float* ln2g      = (const float*)d_in[25];
  const float* ln2b      = (const float*)d_in[26];
  const float* w1        = (const float*)d_in[27];
  const float* b1        = (const float*)d_in[28];
  const float* w2        = (const float*)d_in[29];
  const float* b2        = (const float*)d_in[30];
  const float* lng       = (const float*)d_in[31];
  const float* lnb       = (const float*)d_in[32];
  const float* hw        = (const float*)d_in[33];
  const float* hb        = (const float*)d_in[34];

  // ---- workspace (256B-aligned chunks) ----
  char* wsb = (char*)d_ws;
  auto takeb = [&](long long bytes) { char* p = wsb; wsb += (bytes + 255) & ~255LL; return p; };
  float*    gene  = (float*)takeb((long long)GC * DIMC * 4);
  float*    ctx   = (float*)takeb((long long)BC * DIMC * 4);
  float*    pre   = (float*)takeb(NRC * DIMC * 4);  // later reused as V
  float*    x     = (float*)takeb(NRC * DIMC * 4);
  float*    t1    = (float*)takeb(NRC * DIMC * 4);  // xl / h
  float*    t2    = (float*)takeb(NRC * DIMC * 4);  // xr / Q
  float*    t3    = (float*)takeb(NRC * DIMC * 4);  // K
  float*    t4    = (float*)takeb(NRC * DIMC * 4);  // GAT out / attention O
  float*    elog  = (float*)takeb((long long)BC * ETOT * HC * 4);
  float*    eex   = (float*)takeb((long long)BC * ETOT * HC * 4);
  float*    lmax  = (float*)takeb((long long)BC * GC * HC * 4);
  float*    den   = (float*)takeb((long long)BC * GC * HC * 4);
  float*    dd    = (float*)takeb((long long)GC * NBC * 4);
  float*    diag  = (float*)takeb(GC * 4);
  float*    rmax  = (float*)takeb(GC * 4);
  float*    ksum  = (float*)takeb(NBC * 4);
  float*    ctxm  = (float*)takeb((long long)NBC * DHC * 4);
  float*    gmax  = (float*)takeb(64);
  float*    dinv  = (float*)takeb(GC * 4);
  _Float16* hA    = (_Float16*)takeb(NRC * DIMC * 2);            // f16 A operand
  _Float16* hW    = (_Float16*)takeb((long long)2000000 * 2);    // f16 Wt operand
  _Float16* ffb16 = (_Float16*)takeb(NRC * FFC * 2);             // FFN intermediate

  auto blocks = [](long long n) { return (unsigned)((n + 255) / 256); };
  auto fill = [&](float* p, float v, long long n) {
    fill_kernel<<<blocks(n), 256, 0, stream>>>(p, v, n);
  };
  auto cvt = [&](const float* src, long long lds, _Float16* dst, long long ldd,
                 long long R, long long C, float scale) {
    cvt16_rows_kernel<<<blocks(R * ldd), 256, 0, stream>>>(src, lds, dst, ldd, R, C, scale);
  };
  auto cvtT = [&](const float* src, long long lds, _Float16* dst, long long ldd,
                  long long R, long long Cdst, float scale) {
    cvt16_trans_kernel<<<blocks(Cdst * ldd), 256, 0, stream>>>(src, lds, dst, ldd, R, Cdst, scale);
  };
  auto gemm = [&](const _Float16* A, long long lda, const _Float16* Wt, long long ldw,
                  const float* bias, const float* residual, long long ldr,
                  const float* row_scale, float* C, _Float16* C16, long long ldc,
                  long long M, int N, int K, int act) {
    dim3 grid((unsigned)((M + 127) / 128), (unsigned)((N + 63) / 64));
    wmma_gemm_f16<<<grid, dim3(256), 0, stream>>>(
        A, lda, Wt, ldw, bias, residual, ldr, row_scale, C, C16, ldc, (int)M, N, K, act);
  };

  const float qknorm = powf((float)DHC, -0.25f);
  const float dscale = 0.5f * powf((float)DHC, -0.5f);

  // ---- embedding path ----
  cvt(esm2, DIMC, hA, DIMC, GC, DIMC, 1.0f);
  cvtT(gene_w, DIMC, hW, DIMC, DIMC, DIMC, 1.0f);
  gemm(hA, DIMC, hW, DIMC, gene_b, nullptr, 0, nullptr, gene, nullptr, DIMC, GC, DIMC, DIMC, 0);
  ctx_kernel<<<BC, DIMC, 0, stream>>>(idx, ae_latent, ae_w, ae_b, ctx);
  build_pre_kernel<<<blocks(NRC * DIMC), 256, 0, stream>>>(x_expr, gene, ctx, pre);
  cvt(pre, DIMC, hA, DIMC, NRC, DIMC, 1.0f);
  cvtT(expr_w, DIMC, hW, DIMC, DIMC, DIMC, 1.0f);
  gemm(hA, DIMC, hW, DIMC, expr_b, nullptr, 0, nullptr, x, nullptr, DIMC, NRC, DIMC, DIMC, 0);

  // ---- GATv2 ----
  cvt(x, DIMC, hA, DIMC, NRC, DIMC, 1.0f);
  cvtT(gat_wl, DIMC, hW, DIMC, DIMC, DIMC, 1.0f);
  gemm(hA, DIMC, hW, DIMC, gat_bl, nullptr, 0, nullptr, t1, nullptr, DIMC, NRC, DIMC, DIMC, 0);
  cvtT(gat_wr, DIMC, hW, DIMC, DIMC, DIMC, 1.0f);
  gemm(hA, DIMC, hW, DIMC, gat_br, nullptr, 0, nullptr, t2, nullptr, DIMC, NRC, DIMC, DIMC, 0);
  fill(lmax, -1e30f, (long long)BC * GC * HC);
  fill(den, 0.0f, (long long)BC * GC * HC);
  fill(t4, 0.0f, NRC * DIMC);
  edge_logit_kernel<<<blocks((long long)BC * ETOT * HC), 256, 0, stream>>>(t1, t2, gat_att, eidx, elog, lmax);
  edge_exp_kernel<<<blocks((long long)BC * ETOT * HC), 256, 0, stream>>>(elog, lmax, eidx, eex, den);
  edge_aggr_kernel<<<blocks((long long)BC * ETOT * DIMC), 256, 0, stream>>>(eex, den, t1, eidx, t4);
  add_bias_kernel<<<blocks(NRC * DIMC), 256, 0, stream>>>(t4, gat_bias, x);

  // ---- 2 Performer blocks ----
  const unsigned rows_grid = (unsigned)((NRC + 7) / 8);
  const unsigned g_grid    = (unsigned)((GC + 7) / 8);
  for (int l = 0; l < 2; ++l) {
    layernorm_kernel<<<rows_grid, 256, 0, stream>>>(x, ln1g + l * DIMC, ln1b + l * DIMC, t1, NRC);
    cvt(t1, DIMC, hA, DIMC, NRC, DIMC, 1.0f);
    cvtT(wq + (long long)l * DIMC * DIMC, DIMC, hW, DIMC, DIMC, DIMC, 1.0f);
    gemm(hA, DIMC, hW, DIMC, nullptr, nullptr, 0, nullptr, t2, nullptr, DIMC, NRC, DIMC, DIMC, 0);
    cvtT(wk + (long long)l * DIMC * DIMC, DIMC, hW, DIMC, DIMC, DIMC, 1.0f);
    gemm(hA, DIMC, hW, DIMC, nullptr, nullptr, 0, nullptr, t3, nullptr, DIMC, NRC, DIMC, DIMC, 0);
    cvtT(wv + (long long)l * DIMC * DIMC, DIMC, hW, DIMC, DIMC, DIMC, 1.0f);
    gemm(hA, DIMC, hW, DIMC, nullptr, nullptr, 0, nullptr, pre, nullptr, DIMC, NRC, DIMC, DIMC, 0);

    const float* pj = proj + (long long)l * NBC * DHC;
    for (int b = 0; b < BC; ++b) {
      for (int h = 0; h < HC; ++h) {
        const long long off = ((long long)b * GC) * DIMC + h * DHC;
        const float* Qs = t2 + off;
        const float* Ks = t3 + off;
        const float* Vs = pre + off;
        float*       Os = t4 + off;
        const long long nGN = (long long)GC * NBC;
        // --- key features ---
        cvt(Ks, DIMC, hA, KDH, GC, DHC, qknorm);   // [G][96] padded
        cvt(pj, DHC, hW, KDH, NBC, DHC, 1.0f);     // proj is already [n=m][k=d]
        gemm(hA, KDH, hW, KDH, nullptr, nullptr, 0, nullptr, dd, nullptr, NBC, GC, NBC, KDH, 0);
        diag_kernel<<<(GC + 255) / 256, 256, 0, stream>>>(Ks, diag, dscale);
        fill(gmax, -1e30f, 1);
        reduce_max_kernel<<<1024, 256, 0, stream>>>(dd, nGN, gmax);
        favor_exp_kernel<<<blocks(nGN), 256, 0, stream>>>(dd, diag, gmax, 0);
        fill(ksum, 0.0f, NBC);
        colsum_kernel<<<(GC + 127) / 128, 384, 0, stream>>>(dd, ksum);
        // --- ctxm[NB,DH] = kp^T @ V ---
        cvtT(dd, NBC, hA, KG, GC, NBC, 1.0f);      // hA[m*KG + n] = kp[n][m], padded
        cvtT(Vs, DIMC, hW, KG, GC, DHC, 1.0f);     // hW[d*KG + n] = V[n][d], padded
        gemm(hA, KG, hW, KG, nullptr, nullptr, 0, nullptr, ctxm, nullptr, DHC, NBC, DHC, KG, 0);
        // --- query features (reuse dd) ---
        cvt(Qs, DIMC, hA, KDH, GC, DHC, qknorm);
        cvt(pj, DHC, hW, KDH, NBC, DHC, 1.0f);
        gemm(hA, KDH, hW, KDH, nullptr, nullptr, 0, nullptr, dd, nullptr, NBC, GC, NBC, KDH, 0);
        diag_kernel<<<(GC + 255) / 256, 256, 0, stream>>>(Qs, diag, dscale);
        rowmax_kernel<<<g_grid, 256, 0, stream>>>(dd, rmax);
        favor_exp_kernel<<<blocks(nGN), 256, 0, stream>>>(dd, diag, rmax, 1);
        dinv_kernel<<<g_grid, 256, 0, stream>>>(dd, ksum, dinv);
        // --- O slice = diag(dinv) * (qp @ ctxm) ---
        cvt(dd, NBC, hA, KNB, GC, NBC, 1.0f);      // [G][352] padded
        cvtT(ctxm, DHC, hW, KNB, NBC, DHC, 1.0f);  // hW[d*352 + m], padded
        gemm(hA, KNB, hW, KNB, nullptr, nullptr, 0, dinv, Os, nullptr, DIMC, GC, DHC, KNB, 0);
      }
    }
    // x = x + O @ wo + bo
    cvt(t4, DIMC, hA, DIMC, NRC, DIMC, 1.0f);
    cvtT(wo + (long long)l * DIMC * DIMC, DIMC, hW, DIMC, DIMC, DIMC, 1.0f);
    gemm(hA, DIMC, hW, DIMC, bo + l * DIMC, x, DIMC, nullptr, x, nullptr, DIMC, NRC, DIMC, DIMC, 0);
    // FFN
    layernorm_kernel<<<rows_grid, 256, 0, stream>>>(x, ln2g + l * DIMC, ln2b + l * DIMC, t1, NRC);
    cvt(t1, DIMC, hA, DIMC, NRC, DIMC, 1.0f);
    cvtT(w1 + (long long)l * DIMC * FFC, FFC, hW, DIMC, DIMC, FFC, 1.0f);  // [1280][320]
    gemm(hA, DIMC, hW, DIMC, b1 + (long long)l * FFC, nullptr, 0, nullptr,
         nullptr, ffb16, FFC, NRC, FFC, DIMC, 1);                          // f16 out, gelu
    cvtT(w2 + (long long)l * FFC * DIMC, DIMC, hW, FFC, FFC, DIMC, 1.0f);  // [320][1280]
    gemm(ffb16, FFC, hW, FFC, b2 + l * DIMC, x, DIMC, nullptr, x, nullptr, DIMC, NRC, DIMC, FFC, 0);
  }

  // ---- final LN + head ----
  layernorm_kernel<<<rows_grid, 256, 0, stream>>>(x, lng, lnb, t1, NRC);
  head_kernel<<<rows_grid, 256, 0, stream>>>(t1, hw, hb, (float*)d_out, NRC);
}
